// DeformableTransformerDecoderLayer_Decouple_84095459656062
// MI455X (gfx1250) — matrix operand order repaired and checked by reading, not compiled, not run on hardware
//
#include <hip/hip_runtime.h>

// ---------------- problem constants ----------------
#define BB   16
#define NQn  300
#define DD   512
#define DFF  2048
#define HH   8
#define DHh  64
#define SSt  7680
#define N2   600     // 2*NQ (concat cls+loc)
#define NKP  608     // keys padded to multiple of 16
#define SPITCH (NKP + 8)   // 616 floats: 16B aligned row pitch

typedef __attribute__((ext_vector_type(16))) _Float16 v16h;
typedef __attribute__((ext_vector_type(8)))  float    v8f;

__device__ __forceinline__ v8f wmma_f32_16x16x32(v16h a, v16h b, v8f c) {
  return __builtin_amdgcn_wmma_f32_16x16x32_f16(false, a, false, b, (short)0, c, false, false);
}

// Pack 4 consecutive fragment elements from a float4 (fp32 -> fp16)
__device__ __forceinline__ void put4(v16h& f, int e, float4 v) {
  f[e]     = (_Float16)v.x;
  f[e + 1] = (_Float16)v.y;
  f[e + 2] = (_Float16)v.z;
  f[e + 3] = (_Float16)v.w;
}

// A fragment (16x32 MxK, ISA layout): two 8-float runs at k0+half*8 and k0+16+half*8
__device__ __forceinline__ v16h load_afrag(const float4* __restrict__ row4, int k0, int half) {
  v16h f;
  const int i = (k0 >> 2) + half * 2;
  put4(f, 0,  row4[i]);
  put4(f, 4,  row4[i + 1]);
  put4(f, 8,  row4[i + 4]);
  put4(f, 12, row4[i + 5]);
  return f;
}

// B fragment (32x16 KxN, ISA layout): two 8-float runs at k0+half*8 and k0+16+half*8
__device__ __forceinline__ v16h load_bfrag(const float4* __restrict__ row4, int k0, int half) {
  v16h f;
  const int i = (k0 >> 2) + half * 2;
  put4(f, 0,  row4[i]);
  put4(f, 4,  row4[i + 1]);
  put4(f, 8,  row4[i + 4]);
  put4(f, 12, row4[i + 5]);
  return f;
}

// =====================================================================
// GEMM: C[M,N] = A[M,K] @ W[N,K]^T + bias (torch Linear layout).
// 256 threads = 8 waves; wave tile 16x32 (2 WMMAs/K-step), block tile 32x128.
// Requires M%32==0, N%128==0, K%32==0 (true for all calls here).
// =====================================================================
template<int RELU, int HALF_OUT>
__global__ void __launch_bounds__(256)
gemm_wmma(const float* __restrict__ A, const float* __restrict__ W,
          const float* __restrict__ bias, void* __restrict__ Cv,
          int ldc, int K)
{
  const int wave = threadIdx.x >> 5;
  const int lane = threadIdx.x & 31;
  const int half = lane >> 4;
  const int lr   = lane & 15;
  const int tileM = blockIdx.y * 32  + (wave & 1) * 16;
  const int tileN = blockIdx.x * 128 + (wave >> 1) * 32;

  const float4* __restrict__ Arow  = (const float4*)(A + (size_t)(tileM + lr) * K);
  const float4* __restrict__ Wrow0 = (const float4*)(W + (size_t)(tileN + lr) * K);
  const float4* __restrict__ Wrow1 = (const float4*)(W + (size_t)(tileN + 16 + lr) * K);

  v8f acc0 = {}, acc1 = {};
#pragma unroll 4
  for (int k0 = 0; k0 < K; k0 += 32) {
    __builtin_prefetch(((const float*)Arow) + k0 + 256, 0, 1);
    __builtin_prefetch(((const float*)Wrow0) + k0 + 256, 0, 1);
    __builtin_prefetch(((const float*)Wrow1) + k0 + 256, 0, 1);
    const v16h af = load_afrag(Arow, k0, half);
    const v16h b0 = load_bfrag(Wrow0, k0, half);
    const v16h b1 = load_bfrag(Wrow1, k0, half);
    acc0 = wmma_f32_16x16x32(af, b0, acc0);
    acc1 = wmma_f32_16x16x32(af, b1, acc1);
  }

  const int n0 = tileN + lr;
  const int n1 = tileN + 16 + lr;
  const float bv0 = bias[n0], bv1 = bias[n1];
#pragma unroll
  for (int i = 0; i < 8; ++i) {
    const int m = tileM + half * 8 + i;
    float v0 = acc0[i] + bv0;
    float v1 = acc1[i] + bv1;
    if (RELU) { v0 = fmaxf(v0, 0.f); v1 = fmaxf(v1, 0.f); }
    if (HALF_OUT) {
      ((_Float16*)Cv)[(size_t)m * ldc + n0] = (_Float16)v0;
      ((_Float16*)Cv)[(size_t)m * ldc + n1] = (_Float16)v1;
    } else {
      ((float*)Cv)[(size_t)m * ldc + n0] = v0;
      ((float*)Cv)[(size_t)m * ldc + n1] = v1;
    }
  }
}

// =====================================================================
// Self-attention, one workgroup per (b, h, 16-query tile).
// qkv layout: (B, 600 rows + pad, 1536): [0,512)=Q, [512,1024)=K, [1024,1536)=V
// All loads unconditional from clamped rows; invalid keys masked post-WMMA.
// =====================================================================
__global__ void __launch_bounds__(128)
attn_kernel(const float* __restrict__ qkv, float* __restrict__ out)
{
  __shared__ float S[16][SPITCH];
  const int qt = blockIdx.x, h = blockIdx.y, b = blockIdx.z;
  const int wave = threadIdx.x >> 5, lane = threadIdx.x & 31;
  const int half = lane >> 4, lr = lane & 15;
  const size_t ld = 1536;
  const float* __restrict__ base = qkv + (size_t)b * N2 * ld;

  // Q fragments (DH=64 -> two K=32 chunks); clamp padded rows (outputs unused)
  const int qrow = qt * 16 + lr;
  const int qrc  = qrow < N2 ? qrow : (N2 - 1);
  const float4* Qp = (const float4*)(base + (size_t)qrc * ld + h * DHh);
  const v16h aq0 = load_afrag(Qp, 0, half);
  const v16h aq1 = load_afrag(Qp, 32, half);

  // Phase 1: S = (Q K^T) * 1/sqrt(DH), invalid key columns forced to -1e30
  for (int kt = wave * 16; kt < NKP; kt += 64) {
    const int key = kt + lr;
    const int kc  = key < N2 ? key : (N2 - 1);
    const float4* Kp = (const float4*)(base + (size_t)kc * ld + DD + h * DHh);
    const v16h b0 = load_bfrag(Kp, 0, half);
    const v16h b1 = load_bfrag(Kp, 32, half);
    v8f s = {};
    s = wmma_f32_16x16x32(aq0, b0, s);
    s = wmma_f32_16x16x32(aq1, b1, s);
    const int col = kt + lr;              // C layout: N = lane&15
    const bool bad = col >= N2;
#pragma unroll
    for (int i = 0; i < 8; ++i) {
      float v = s[i] * 0.125f;
      S[half * 8 + i][col] = bad ? -1e30f : v;   // M = i + half*8
    }
  }
  __syncthreads();

  // Phase 2: softmax per row, 8 threads/row
  {
    const int row = threadIdx.x >> 3, sub = threadIdx.x & 7;
    float mx = -1e30f;
    for (int c = sub; c < NKP; c += 8) mx = fmaxf(mx, S[row][c]);
#pragma unroll
    for (int m = 1; m < 8; m <<= 1) mx = fmaxf(mx, __shfl_xor(mx, m, 32));
    float sum = 0.f;
    for (int c = sub; c < NKP; c += 8) {
      float e = __expf(S[row][c] - mx);
      S[row][c] = e; sum += e;
    }
#pragma unroll
    for (int m = 1; m < 8; m <<= 1) sum += __shfl_xor(sum, m, 32);
    const float inv = 1.f / sum;
    for (int c = sub; c < NKP; c += 8) S[row][c] *= inv;
  }
  __syncthreads();

  // Phase 3: O = P @ V; wave handles dh columns [wave*16, wave*16+16)
  const int dh = wave * 16 + lr;
  const float4* Srow = (const float4*)(&S[lr][0]);   // pitch 616f: 16B aligned
  v8f oacc = {};
  for (int kt = 0; kt < NKP; kt += 32) {
    v16h ap;
    {
      const int i = (kt >> 2) + half * 2;
      put4(ap, 0,  Srow[i]);
      put4(ap, 4,  Srow[i + 1]);
      put4(ap, 8,  Srow[i + 4]);
      put4(ap, 12, Srow[i + 5]);
    }
    v16h bv;
#pragma unroll
    for (int e = 0; e < 16; ++e) {
      int key = kt + half * 8 + e + (e >= 8 ? 8 : 0);
      if (key >= N2) key = N2 - 1;        // P==0 there
      bv[e] = (_Float16)base[(size_t)key * ld + 2 * DD + h * DHh + dh];
    }
    oacc = wmma_f32_16x16x32(ap, bv, oacc);
  }
#pragma unroll
  for (int i = 0; i < 8; ++i) {
    const int q = qt * 16 + half * 8 + i;
    if (q < N2)
      out[((size_t)b * N2 + q) * DD + h * DHh + dh] = oacc[i];
  }
}

// =====================================================================
// Elementwise / LN / softmax / sampling kernels
// =====================================================================
__global__ void build_cat(const float* __restrict__ tc, const float* __restrict__ tl,
                          const float* __restrict__ pc, const float* __restrict__ pl,
                          float* __restrict__ tgt_cat, float* __restrict__ q, int total)
{
  int i = blockIdx.x * blockDim.x + threadIdx.x;
  if (i >= total) return;
  int c = i % DD, n = (i / DD) % N2, b = i / (DD * N2);
  float t, p;
  if (n < NQn) { size_t j = ((size_t)b * NQn + n) * DD + c;         t = tc[j]; p = pc[j]; }
  else         { size_t j = ((size_t)b * NQn + (n - NQn)) * DD + c; t = tl[j]; p = pl[j]; }
  tgt_cat[i] = t;
  q[i] = t + p;
}

__global__ void extract_add(const float* __restrict__ tgt2, const float* __restrict__ qpos,
                            int off600, float* __restrict__ qc, float* __restrict__ tpart, int total)
{
  int i = blockIdx.x * blockDim.x + threadIdx.x;
  if (i >= total) return;
  int c = i % DD, n = (i / DD) % NQn, b = i / (DD * NQn);
  float t = tgt2[((size_t)b * N2 + off600 + n) * DD + c];
  tpart[i] = t;
  qc[i] = t + qpos[i];
}

// y = LayerNorm(x + r) * g + b ; one wave per 512-wide row
__global__ void __launch_bounds__(256)
add_ln(const float* __restrict__ x, const float* __restrict__ r,
       const float* __restrict__ g, const float* __restrict__ bta,
       float* __restrict__ out, int rows)
{
  const int wave = threadIdx.x >> 5, lane = threadIdx.x & 31;
  const int row = blockIdx.x * 8 + wave;
  if (row >= rows) return;
  const float4* xp = (const float4*)(x + (size_t)row * DD);
  const float4* rp = (const float4*)(r + (size_t)row * DD);
  float4 v[4]; float s = 0.f;
#pragma unroll
  for (int i = 0; i < 4; ++i) {
    float4 a = xp[lane + 32 * i], d = rp[lane + 32 * i];
    v[i].x = a.x + d.x; v[i].y = a.y + d.y; v[i].z = a.z + d.z; v[i].w = a.w + d.w;
    s += v[i].x + v[i].y + v[i].z + v[i].w;
  }
#pragma unroll
  for (int m = 1; m < 32; m <<= 1) s += __shfl_xor(s, m, 32);
  const float mean = s * (1.f / DD);
  float var = 0.f;
#pragma unroll
  for (int i = 0; i < 4; ++i) {
    float dx = v[i].x - mean, dy = v[i].y - mean, dz = v[i].z - mean, dw = v[i].w - mean;
    var += dx * dx + dy * dy + dz * dz + dw * dw;
  }
#pragma unroll
  for (int m = 1; m < 32; m <<= 1) var += __shfl_xor(var, m, 32);
  const float inv = rsqrtf(var * (1.f / DD) + 1e-5f);
  const float4* g4 = (const float4*)g;
  const float4* b4 = (const float4*)bta;
  float4* op = (float4*)(out + (size_t)row * DD);
#pragma unroll
  for (int i = 0; i < 4; ++i) {
    const int c = lane + 32 * i;
    float4 gg = g4[c], bb = b4[c], o;
    o.x = (v[i].x - mean) * inv * gg.x + bb.x;
    o.y = (v[i].y - mean) * inv * gg.y + bb.y;
    o.z = (v[i].z - mean) * inv * gg.z + bb.z;
    o.w = (v[i].w - mean) * inv * gg.w + bb.w;
    op[c] = o;
  }
}

// softmax over 16 (L*P) values per (b,q,h); writes normalized aw to d_out
__global__ void aw_softmax(const float* __restrict__ raw, float* __restrict__ out, int n)
{
  int i = blockIdx.x * blockDim.x + threadIdx.x;
  if (i >= n) return;
  const float4* p4 = (const float4*)(raw + (size_t)i * 16);
  float v[16], mx = -1e30f;
#pragma unroll
  for (int j = 0; j < 4; ++j) {
    float4 t = p4[j];
    v[4*j] = t.x; v[4*j+1] = t.y; v[4*j+2] = t.z; v[4*j+3] = t.w;
  }
#pragma unroll
  for (int j = 0; j < 16; ++j) mx = fmaxf(mx, v[j]);
  float sum = 0.f;
#pragma unroll
  for (int j = 0; j < 16; ++j) { v[j] = __expf(v[j] - mx); sum += v[j]; }
  const float inv = 1.f / sum;
  float4* o4 = (float4*)(out + (size_t)i * 16);
#pragma unroll
  for (int j = 0; j < 4; ++j) {
    float4 t; t.x = v[4*j] * inv; t.y = v[4*j+1] * inv; t.z = v[4*j+2] * inv; t.w = v[4*j+3] * inv;
    o4[j] = t;
  }
}

// MS-deform bilinear sampling: one wave per (b,q,h); lane covers 2 of 64 dh.
__global__ void __launch_bounds__(256)
msdeform_sample(const _Float16* __restrict__ value,  // (B, S, 512) f16, col = h*64+dh
                const float* __restrict__ off,       // (B,300,8,4,4)
                const float* __restrict__ awn,       // normalized weights
                const float* __restrict__ ref,       // (B,300,4,1)
                float* __restrict__ sampled,         // (B,300,512)
                float* __restrict__ sl)              // (B,300,8,4,4)
{
  const int wave = threadIdx.x >> 5, lane = threadIdx.x & 31;
  const int gid = blockIdx.x * 8 + wave;
  if (gid >= BB * NQn * HH) return;
  const int h  = gid & 7;
  const int bq = gid >> 3;
  const int b  = bq / NQn;

  const float* offp = off + (size_t)bq * 128 + h * 16;
  const float* awp  = awn + (size_t)bq * 128 + h * 16;
  const float* refp = ref + (size_t)bq * 4;

  const int starts[4] = {0, 4096, 6144, 7168};
  const int lv[4]     = {4096, 2048, 1024, 512};

  const int dh0 = lane * 2, dh1 = lane * 2 + 1;
  float acc0 = 0.f, acc1 = 0.f;

#pragma unroll
  for (int l = 0; l < 4; ++l) {
    const float rr = refp[l];
    const float tn = (float)lv[l];
    const _Float16* vb = value + ((size_t)b * SSt + starts[l]) * DD + h * DHh;
#pragma unroll
    for (int p = 0; p < 4; ++p) {
      const float o   = offp[l * 4 + p];
      const float loc = rr + o / tn;
      const float x   = loc * tn - 0.5f;
      const float x0  = floorf(x);
      const float w1  = x - x0;
      const int   i0  = (int)x0;
      const float aw  = awp[l * 4 + p];
      float s00 = 0.f, s01 = 0.f, s10 = 0.f, s11 = 0.f;
      if (i0 >= 0 && i0 < lv[l]) {
        const _Float16* vp = vb + (size_t)i0 * DD;
        s00 = (float)vp[dh0]; s01 = (float)vp[dh1];
      }
      const int i1 = i0 + 1;
      if (i1 >= 0 && i1 < lv[l]) {
        const _Float16* vp = vb + (size_t)i1 * DD;
        s10 = (float)vp[dh0]; s11 = (float)vp[dh1];
      }
      acc0 += aw * (s00 * (1.f - w1) + s10 * w1);
      acc1 += aw * (s01 * (1.f - w1) + s11 * w1);
    }
  }
  sampled[(size_t)bq * DD + h * DHh + dh0] = acc0;
  sampled[(size_t)bq * DD + h * DHh + dh1] = acc1;

  if (lane < 16) {
    const int l = lane >> 2, p = lane & 3;
    const float loc = refp[l] + offp[l * 4 + p] / (float)lv[l];
    sl[(size_t)bq * 128 + h * 16 + lane] = loc;
  }
}

// =====================================================================
// Host orchestration
// =====================================================================
extern "C" void kernel_launch(void* const* d_in, const int* in_sizes, int n_in,
                              void* d_out, int out_size, void* d_ws, size_t ws_size,
                              hipStream_t stream)
{
  (void)in_sizes; (void)n_in; (void)out_size; (void)ws_size;

  const float* tgt_cls  = (const float*)d_in[0];
  const float* tgt_loc  = (const float*)d_in[1];
  const float* qpos_cls = (const float*)d_in[2];
  const float* qpos_loc = (const float*)d_in[3];
  const float* ref_cls  = (const float*)d_in[4];
  const float* ref_loc  = (const float*)d_in[5];
  const float* src      = (const float*)d_in[6];
  const float* sa_in_w  = (const float*)d_in[9];
  const float* sa_in_b  = (const float*)d_in[10];
  const float* sa_out_w = (const float*)d_in[11];
  const float* sa_out_b = (const float*)d_in[12];
  const float* n1g = (const float*)d_in[29]; const float* n1b = (const float*)d_in[30];
  const float* n2g = (const float*)d_in[31]; const float* n2b = (const float*)d_in[32];
  const float* n3g = (const float*)d_in[33]; const float* n3b = (const float*)d_in[34];

  const size_t ROWS6   = (size_t)BB * N2;            // 9600
  const size_t SZ600   = ROWS6 * DD;
  const size_t SZQKV   = (size_t)BB * NKP * 1536;
  const size_t SZ300   = (size_t)BB * NQn * DD;
  const size_t SZ128   = (size_t)BB * NQn * 128;
  const size_t SZFFN   = (size_t)BB * NQn * DFF;
  const size_t SZVALF  = ((size_t)BB * SSt * DD + 1) / 2;

  float* wsf     = (float*)d_ws;
  float* tgt_cat = wsf;                 float* qbuf   = tgt_cat + SZ600;
  float* qkv     = qbuf + SZ600;        float* attn_o = qkv + SZQKV;
  float* sa_o    = attn_o + SZ600;      float* tgt2   = sa_o + SZ600;
  _Float16* value = (_Float16*)(tgt2 + SZ600);
  float* qc      = (float*)(tgt2 + SZ600) + 2 * SZVALF;
  float* offb    = qc + SZ300;          float* awraw  = offb + SZ128;
  float* sampled = awraw + SZ128;       float* c2     = sampled + SZ300;
  float* tpart   = c2 + SZ300;          float* tcbuf  = tpart + SZ300;
  float* ffnh    = tcbuf + SZ300;       float* ffnf   = ffnh + SZFFN;

  float* out = (float*)d_out;
  float* out_tc  = out;
  float* out_tl  = out + SZ300;
  float* out_slc = out + 2 * SZ300;
  float* out_sll = out_slc + SZ128;
  float* out_awc = out_sll + SZ128;
  float* out_awl = out_awc + SZ128;

  auto gemm = [&](const float* A, const float* W, const float* bias, void* C,
                  int ldc, int M, int N, int K, int relu, int halfout) {
    dim3 grid(N / 128, M / 32);
    if (halfout)    gemm_wmma<0, 1><<<grid, 256, 0, stream>>>(A, W, bias, C, ldc, K);
    else if (relu)  gemm_wmma<1, 0><<<grid, 256, 0, stream>>>(A, W, bias, C, ldc, K);
    else            gemm_wmma<0, 0><<<grid, 256, 0, stream>>>(A, W, bias, C, ldc, K);
  };

  // 1) concat + positional add
  build_cat<<<(int)((SZ600 + 255) / 256), 256, 0, stream>>>(
      tgt_cls, tgt_loc, qpos_cls, qpos_loc, tgt_cat, qbuf, (int)SZ600);
  // 2) QKV projection (Q,K from q; V from tgt)
  gemm(qbuf,    sa_in_w,             sa_in_b,        qkv,        1536, (int)ROWS6, 1024, DD, 0, 0);
  gemm(tgt_cat, sa_in_w + 1024 * DD, sa_in_b + 1024, qkv + 1024, 1536, (int)ROWS6,  512, DD, 0, 0);
  // 3) self-attention
  {
    dim3 grid(NKP / 16, HH, BB);
    attn_kernel<<<grid, 128, 0, stream>>>(qkv, attn_o);
  }
  // 4) out-proj + residual LN (norm2)
  gemm(attn_o, sa_out_w, sa_out_b, sa_o, DD, (int)ROWS6, DD, DD, 0, 0);
  add_ln<<<(int)(ROWS6 / 8), 256, 0, stream>>>(tgt_cat, sa_o, n2g, n2b, tgt2, (int)ROWS6);

  auto run_branch = [&](int off600, const float* qpos, const float* ref, int pi,
                        const float* l1, const float* b1, const float* l2, const float* b2,
                        float* o_tc, float* o_sl, float* o_aw) {
    const float* vw  = (const float*)d_in[pi + 0];
    const float* vb  = (const float*)d_in[pi + 1];
    const float* ofw = (const float*)d_in[pi + 2];
    const float* ofb = (const float*)d_in[pi + 3];
    const float* aww = (const float*)d_in[pi + 4];
    const float* awb = (const float*)d_in[pi + 5];
    const float* cow = (const float*)d_in[pi + 6];
    const float* cob = (const float*)d_in[pi + 7];

    const int M300 = BB * NQn;
    extract_add<<<(int)((SZ300 + 255) / 256), 256, 0, stream>>>(
        tgt2, qpos, off600, qc, tpart, (int)SZ300);
    gemm(src, vw, vb, value, DD, BB * SSt, DD, DD, 0, 1);      // value proj, f16 out
    gemm(qc, ofw, ofb, offb,  128, M300, 128, DD, 0, 0);       // sampling offsets
    gemm(qc, aww, awb, awraw, 128, M300, 128, DD, 0, 0);       // attn weight logits
    aw_softmax<<<(M300 * HH + 255) / 256, 256, 0, stream>>>(awraw, o_aw, M300 * HH);
    msdeform_sample<<<(M300 * HH) / 8, 256, 0, stream>>>(
        value, offb, o_aw, ref, sampled, o_sl);
    gemm(sampled, cow, cob, c2, DD, M300, DD, DD, 0, 0);       // CA out-proj
    add_ln<<<M300 / 8, 256, 0, stream>>>(tpart, c2, n1g, n1b, tcbuf, M300);
    gemm(tcbuf, l1, b1, ffnh, DFF, M300, DFF, DD, 1, 0);       // FFN up + relu
    gemm(ffnh,  l2, b2, ffnf, DD,  M300, DD, DFF, 0, 0);       // FFN down
    add_ln<<<M300 / 8, 256, 0, stream>>>(tcbuf, ffnf, n3g, n3b, o_tc, M300);
  };

  run_branch(0,   qpos_cls, ref_cls, 13,
             (const float*)d_in[35], (const float*)d_in[36],
             (const float*)d_in[37], (const float*)d_in[38],
             out_tc, out_slc, out_awc);
  run_branch(NQn, qpos_loc, ref_loc, 21,
             (const float*)d_in[39], (const float*)d_in[40],
             (const float*)d_in[41], (const float*)d_in[42],
             out_tl, out_sll, out_awl);
}
